// Attention_4423816315129
// MI455X (gfx1250) — matrix operand order repaired and checked by reading, not compiled, not run on hardware
//
#include <hip/hip_runtime.h>
#include <hip/hip_bf16.h>

// ---------------------------------------------------------------------------
// CDNA5 (gfx1250) WMMA attention pipeline.
// q/k/v + RoPE GEMM -> per-head 32x32 attention -> output projection.
// All matmuls via v_wmma_f32_16x16x32_f16 (f16 inputs, f32 accumulate).
// ---------------------------------------------------------------------------

typedef __attribute__((ext_vector_type(16))) _Float16 v16h;
typedef __attribute__((ext_vector_type(8)))  float    v8f;
typedef __attribute__((ext_vector_type(4)))  _Float16 h4;

union Frag16 { v16h v; _Float16 h[16]; float4 f4[2]; };
union Acc8   { v8f  v; float f[8]; };

#define WMMA_F16(A, B, C) \
  __builtin_amdgcn_wmma_f32_16x16x32_f16(false, (A), false, (B), (short)0, (C), false, false)

static constexpr int BB   = 2;     // batch
static constexpr int S    = 1024;  // spatial positions (hw)
static constexpr int NT   = 32;    // temporal tokens
static constexpr int DIMX = 512;   // model dim
static constexpr int H    = 8;     // heads
static constexpr int DH   = 64;    // head dim
static constexpr int HID  = 512;   // H*DH
static constexpr int NKV  = 1536;  // q(512) + k(512) + v(512)
static constexpr int MTOT = BB * S * NT;  // 65536 tokens

// Workspace layout (units: _Float16 elements). Total = 256 MB.
static constexpr size_t WS_Q = 0;
static constexpr size_t WS_K = (size_t)MTOT * HID;      // after q
static constexpr size_t WS_V = WS_K * 2;                // v stored transposed (d-major)
static constexpr size_t WS_O = WS_K * 3;                // attention output (token, hidden)

// ---------------------------------------------------------------------------
// Kernel 1: [q|k|v] = x @ [w_q | w_kv], fused scale + RoPE on q/k.
// Block tile 128(M) x 128(N), K-step 32. 8 waves, each 32x64 subtile
// (2x4 WMMA accumulators). Writes f16 to workspace in head-major layout;
// v is written transposed so kernel 2's B fragments are contiguous.
// ---------------------------------------------------------------------------
__global__ __launch_bounds__(256)
void qkv_rope_kernel(const float* __restrict__ x,
                     const float* __restrict__ w_q,
                     const float* __restrict__ w_kv,
                     const float* __restrict__ inv_freq,
                     _Float16* __restrict__ ws)
{
  __shared__ __align__(16) _Float16 lA[128 * 40];  // x tile, row-major [m][k], pad 40
  __shared__ __align__(16) _Float16 lB[128 * 40];  // w tile, transposed [n][k], pad 40

  const int tid   = threadIdx.x;
  const int mBase = blockIdx.x * 128;
  const int nBase = blockIdx.y * 128;
  const int sec   = nBase >> 9;  // 0=q, 1=k, 2=v (uniform per block; 128 | 512)

  const float* wsrc    = (sec == 0) ? w_q : w_kv;
  const int    wstride = (sec == 0) ? 512 : 1024;
  const int    cOff    = (sec == 0) ? nBase : (nBase - 512);

  const int  lane = tid & 31;
  const int  wid  = tid >> 5;
  const int  l16  = lane & 15;
  const bool hi   = lane >= 16;
  const int  moff = (wid & 3) * 32;   // wave M offset within block tile
  const int  noff = (wid >> 2) * 64;  // wave N offset within block tile

  Acc8 acc[2][4];
#pragma unroll
  for (int i = 0; i < 2; ++i)
#pragma unroll
    for (int j = 0; j < 4; ++j)
#pragma unroll
      for (int e = 0; e < 8; ++e) acc[i][j].f[e] = 0.0f;

  for (int kb = 0; kb < DIMX; kb += 32) {
    // Stage A: x[mBase..+127][kb..+31] f32 -> f16 LDS (row-major).
#pragma unroll
    for (int i = 0; i < 4; ++i) {
      int idx = tid + i * 256;            // 1024 float4 chunks
      int row = idx >> 3;
      int c4  = (idx & 7) * 4;
      float4 f = *(const float4*)(x + (size_t)(mBase + row) * DIMX + kb + c4);
      h4 hh; hh.x = (_Float16)f.x; hh.y = (_Float16)f.y;
             hh.z = (_Float16)f.z; hh.w = (_Float16)f.w;
      *(h4*)(lA + row * 40 + c4) = hh;
    }
    // Stage B transposed: w[kb..+31][cOff..+127] -> lB[n][k].
#pragma unroll
    for (int i = 0; i < 4; ++i) {
      int idx = tid + i * 256;
      int kr  = idx >> 5;
      int c4  = (idx & 31) * 4;
      float4 f = *(const float4*)(wsrc + (size_t)(kb + kr) * wstride + cOff + c4);
      lB[(c4 + 0) * 40 + kr] = (_Float16)f.x;
      lB[(c4 + 1) * 40 + kr] = (_Float16)f.y;
      lB[(c4 + 2) * 40 + kr] = (_Float16)f.z;
      lB[(c4 + 3) * 40 + kr] = (_Float16)f.w;
    }
    __syncthreads();

    // A fragment: lane-half K split {0-7,16-23} / {8-15,24-31}.
    const int ka = hi ? 8 : 0;
    // B fragment: lane = column, lane-half K split {0-7,16-23} / {8-15,24-31}.
    const int kbf = hi ? 8 : 0;

    Frag16 aF[2], bF[4];
#pragma unroll
    for (int mt = 0; mt < 2; ++mt) {
      const _Float16* p = lA + (moff + mt * 16 + l16) * 40 + ka;
      aF[mt].f4[0] = *(const float4*)p;
      aF[mt].f4[1] = *(const float4*)(p + 16);
    }
#pragma unroll
    for (int nt = 0; nt < 4; ++nt) {
      const _Float16* p = lB + (noff + nt * 16 + l16) * 40 + kbf;
      bF[nt].f4[0] = *(const float4*)p;
      bF[nt].f4[1] = *(const float4*)(p + 16);
    }
#pragma unroll
    for (int mt = 0; mt < 2; ++mt)
#pragma unroll
      for (int nt = 0; nt < 4; ++nt)
        acc[mt][nt].v = WMMA_F16(aF[mt].v, bF[nt].v, acc[mt][nt].v);
    __syncthreads();
  }

  _Float16* wsq = ws + WS_Q;
  _Float16* wsk = ws + WS_K;
  _Float16* wsv = ws + WS_V;

  // Epilogue: C element (r, lane) -> M = r + 8*hi, N = l16 (per 16x16 tile).
#pragma unroll
  for (int mt = 0; mt < 2; ++mt)
#pragma unroll
    for (int nt = 0; nt < 4; ++nt)
#pragma unroll
      for (int r = 0; r < 8; ++r) {
        int Ml  = moff + mt * 16 + r + (hi ? 8 : 0);
        int m   = mBase + Ml;
        int bs  = m >> 5;   // (b*S + s) index
        int pos = m & 31;   // temporal position (RoPE position)
        int c   = nBase + noff + nt * 16 + l16;  // 0..1535
        float val = acc[mt][nt].f[r];
        if (sec < 2) {
          // q or k: scale (q only) + interleaved RoPE. Pair partner lives in
          // lane^1 (same row M, adjacent column) -> wave shuffle.
          int cc = c - sec * 512;
          int hh = cc >> 6, dd = cc & 63;
          if (sec == 0) val *= 0.125f;  // DIM_HEAD^-0.5
          float prt = __shfl_xor(val, 1);
          float fr  = (float)pos * inv_freq[dd >> 1];
          float cs  = __cosf(fr), sn = __sinf(fr);
          float ov  = (dd & 1) ? fmaf(val, cs,  prt * sn)
                               : fmaf(val, cs, -prt * sn);
          _Float16* dst = sec ? wsk : wsq;
          dst[(((size_t)bs * H + hh) * NT + pos) * DH + dd] = (_Float16)ov;
        } else {
          // v: store transposed (b,s,h,d,n) so attn@v B-fragments are contiguous.
          int cc = c - 1024;
          int hh = cc >> 6, dd = cc & 63;
          wsv[(((size_t)bs * H + hh) * DH + dd) * NT + pos] = (_Float16)val;
        }
      }
}

// ---------------------------------------------------------------------------
// Kernel 2: per-head 32x32 attention. One block per (b,s) pair, one wave per
// head (focus-mask branch is block-uniform). sim = q k^T (K=64) -> +bias ->
// softmax (or exact one-hot for focus batches) -> attn v (K=32).
// ---------------------------------------------------------------------------
__global__ __launch_bounds__(256)
void attn_kernel(const float* __restrict__ pos_bias,
                 const int* __restrict__ focus_mask,
                 const _Float16* __restrict__ ws,
                 _Float16* __restrict__ wso)
{
  __shared__ __align__(16) float    sS[H * 32 * 33];  // sim + bias (f32)
  __shared__ __align__(16) _Float16 sP[H * 32 * 40];  // attn probs (f16)

  const int  tid  = threadIdx.x;
  const int  bs   = blockIdx.x;        // 0..2047 (b*S + s)
  const int  b    = bs >> 10;
  const bool focus = focus_mask[b] != 0;
  const int  lane = tid & 31;
  const int  wid  = tid >> 5;          // head
  const int  l16  = lane & 15;
  const bool hiL  = lane >= 16;
  const int  h    = wid;

  const _Float16* qh = ws + WS_Q + ((size_t)bs * H + h) * NT * DH;  // [n][d]
  const _Float16* kh = ws + WS_K + ((size_t)bs * H + h) * NT * DH;  // [n][d]
  const _Float16* vh = ws + WS_V + ((size_t)bs * H + h) * DH * NT;  // [d][n]

  // ---- Phase 1: sim = q @ k^T, 32x32, K=64 (2 K-steps of 32) ----
  Acc8 sim[2][2];
#pragma unroll
  for (int i = 0; i < 2; ++i)
#pragma unroll
    for (int j = 0; j < 2; ++j)
#pragma unroll
      for (int e = 0; e < 8; ++e) sim[i][j].f[e] = 0.0f;

#pragma unroll
  for (int ks = 0; ks < 2; ++ks) {
    const int ka  = ks * 32 + (hiL ? 8 : 0);   // A fragment K offsets
    const int kbf = ks * 32 + (hiL ? 8 : 0);   // B fragment K offsets
    Frag16 aQ[2], bK[2];
#pragma unroll
    for (int mt = 0; mt < 2; ++mt) {
      const _Float16* p = qh + (mt * 16 + l16) * DH + ka;
      aQ[mt].f4[0] = *(const float4*)p;
      aQ[mt].f4[1] = *(const float4*)(p + 16);
    }
#pragma unroll
    for (int nt = 0; nt < 2; ++nt) {
      // B = k^T: column j of B is row j of k (contiguous in d). 
      const _Float16* p = kh + (nt * 16 + l16) * DH + kbf;
      bK[nt].f4[0] = *(const float4*)p;
      bK[nt].f4[1] = *(const float4*)(p + 16);
    }
#pragma unroll
    for (int mt = 0; mt < 2; ++mt)
#pragma unroll
      for (int nt = 0; nt < 2; ++nt)
        sim[mt][nt].v = WMMA_F16(aQ[mt].v, bK[nt].v, sim[mt][nt].v);
  }

  // sim + pos_bias -> LDS
  {
    float* sSh = sS + h * 32 * 33;
    const float* pbh = pos_bias + (size_t)h * NT * NT;
#pragma unroll
    for (int mt = 0; mt < 2; ++mt)
#pragma unroll
      for (int nt = 0; nt < 2; ++nt)
#pragma unroll
        for (int r = 0; r < 8; ++r) {
          int M = mt * 16 + r + (hiL ? 8 : 0);
          int N = nt * 16 + l16;
          sSh[M * 33 + N] = sim[mt][nt].f[r] + pbh[M * 32 + N];
        }
  }
  __syncthreads();

  // ---- Phase 2: softmax, one thread per (head,row) ----
  {
    const int  rh  = tid >> 5;   // head
    const int  row = tid & 31;
    const float*    rp = sS + rh * 32 * 33 + row * 33;
    _Float16* pr = sP + rh * 32 * 40 + row * 40;
    if (focus) {
      // mask = eye: attend only to self -> exact one-hot probabilities
#pragma unroll
      for (int j = 0; j < 32; ++j)
        pr[j] = (_Float16)((j == row) ? 1.0f : 0.0f);
    } else {
      float mx = -3.0e38f;
#pragma unroll
      for (int j = 0; j < 32; ++j) mx = fmaxf(mx, rp[j]);
      float ev[32];
      float sum = 0.0f;
#pragma unroll
      for (int j = 0; j < 32; ++j) { ev[j] = __expf(rp[j] - mx); sum += ev[j]; }
      float inv = 1.0f / sum;
#pragma unroll
      for (int j = 0; j < 32; ++j) pr[j] = (_Float16)(ev[j] * inv);
    }
  }
  __syncthreads();

  // ---- Phase 3: o = attn @ v, 32x64, K=32 (single K-step) ----
  Acc8 o[2][4];
#pragma unroll
  for (int i = 0; i < 2; ++i)
#pragma unroll
    for (int j = 0; j < 4; ++j)
#pragma unroll
      for (int e = 0; e < 8; ++e) o[i][j].f[e] = 0.0f;

  {
    const int ka  = hiL ? 8 : 0;
    const int kbf = hiL ? 8 : 0;
    Frag16 aP[2], bV[4];
#pragma unroll
    for (int mt = 0; mt < 2; ++mt) {
      const _Float16* p = sP + h * 32 * 40 + (mt * 16 + l16) * 40 + ka;
      aP[mt].f4[0] = *(const float4*)p;
      aP[mt].f4[1] = *(const float4*)(p + 16);
    }
#pragma unroll
    for (int nt = 0; nt < 4; ++nt) {
      // B = v (32 keys x 64 dims); vh is [d][n] so column d is contiguous in n.
      const _Float16* p = vh + (nt * 16 + l16) * NT + kbf;
      bV[nt].f4[0] = *(const float4*)p;
      bV[nt].f4[1] = *(const float4*)(p + 16);
    }
#pragma unroll
    for (int mt = 0; mt < 2; ++mt)
#pragma unroll
      for (int nt = 0; nt < 4; ++nt)
        o[mt][nt].v = WMMA_F16(aP[mt].v, bV[nt].v, o[mt][nt].v);
  }

  // Store attention output (token, hidden) f16 for the final GEMM.
#pragma unroll
  for (int mt = 0; mt < 2; ++mt)
#pragma unroll
    for (int nt = 0; nt < 4; ++nt)
#pragma unroll
      for (int r = 0; r < 8; ++r) {
        int qrow = mt * 16 + r + (hiL ? 8 : 0);
        int dcol = nt * 16 + l16;
        wso[((size_t)bs * NT + qrow) * HID + h * DH + dcol] =
            (_Float16)o[mt][nt].f[r];
      }
}

// ---------------------------------------------------------------------------
// Kernel 3: out = attn_out @ w_out. Same tiling as kernel 1; A already f16.
// ---------------------------------------------------------------------------
__global__ __launch_bounds__(256)
void out_proj_kernel(const _Float16* __restrict__ a,
                     const float* __restrict__ w_out,
                     float* __restrict__ out)
{
  __shared__ __align__(16) _Float16 lA[128 * 40];
  __shared__ __align__(16) _Float16 lB[128 * 40];

  const int tid   = threadIdx.x;
  const int mBase = blockIdx.x * 128;
  const int nBase = blockIdx.y * 128;

  const int  lane = tid & 31;
  const int  wid  = tid >> 5;
  const int  l16  = lane & 15;
  const bool hi   = lane >= 16;
  const int  moff = (wid & 3) * 32;
  const int  noff = (wid >> 2) * 64;

  Acc8 acc[2][4];
#pragma unroll
  for (int i = 0; i < 2; ++i)
#pragma unroll
    for (int j = 0; j < 4; ++j)
#pragma unroll
      for (int e = 0; e < 8; ++e) acc[i][j].f[e] = 0.0f;

  for (int kb = 0; kb < HID; kb += 32) {
    // Stage A (already f16): straight 16B copies.
#pragma unroll
    for (int i = 0; i < 2; ++i) {
      int chunk = tid + i * 256;          // 512 chunks of 8 halves
      int row = chunk >> 2;
      int c8  = (chunk & 3) * 8;
      *(float4*)(lA + row * 40 + c8) =
          *(const float4*)(a + (size_t)(mBase + row) * HID + kb + c8);
    }
    // Stage B: w_out f32 -> f16, transposed [n][k].
#pragma unroll
    for (int i = 0; i < 4; ++i) {
      int idx = tid + i * 256;
      int kr  = idx >> 5;
      int c4  = (idx & 31) * 4;
      float4 f = *(const float4*)(w_out + (size_t)(kb + kr) * HID + nBase + c4);
      lB[(c4 + 0) * 40 + kr] = (_Float16)f.x;
      lB[(c4 + 1) * 40 + kr] = (_Float16)f.y;
      lB[(c4 + 2) * 40 + kr] = (_Float16)f.z;
      lB[(c4 + 3) * 40 + kr] = (_Float16)f.w;
    }
    __syncthreads();

    const int ka  = hi ? 8 : 0;
    const int kbf = hi ? 8 : 0;
    Frag16 aF[2], bF[4];
#pragma unroll
    for (int mt = 0; mt < 2; ++mt) {
      const _Float16* p = lA + (moff + mt * 16 + l16) * 40 + ka;
      aF[mt].f4[0] = *(const float4*)p;
      aF[mt].f4[1] = *(const float4*)(p + 16);
    }
#pragma unroll
    for (int nt = 0; nt < 4; ++nt) {
      const _Float16* p = lB + (noff + nt * 16 + l16) * 40 + kbf;
      bF[nt].f4[0] = *(const float4*)p;
      bF[nt].f4[1] = *(const float4*)(p + 16);
    }
#pragma unroll
    for (int mt = 0; mt < 2; ++mt)
#pragma unroll
      for (int nt = 0; nt < 4; ++nt)
        acc[mt][nt].v = WMMA_F16(aF[mt].v, bF[nt].v, acc[mt][nt].v);
    __syncthreads();
  }

#pragma unroll
  for (int mt = 0; mt < 2; ++mt)
#pragma unroll
    for (int nt = 0; nt < 4; ++nt)
#pragma unroll
      for (int r = 0; r < 8; ++r) {
        int Ml = moff + mt * 16 + r + (hi ? 8 : 0);
        int c  = nBase + noff + nt * 16 + l16;
        out[(size_t)(mBase + Ml) * HID + c] = acc[mt][nt].f[r];
      }
}

// ---------------------------------------------------------------------------
extern "C" void kernel_launch(void* const* d_in, const int* in_sizes, int n_in,
                              void* d_out, int out_size, void* d_ws, size_t ws_size,
                              hipStream_t stream)
{
  (void)in_sizes; (void)n_in; (void)out_size; (void)ws_size;

  const float* x        = (const float*)d_in[0];
  const float* pos_bias = (const float*)d_in[1];
  const int*   focus    = (const int*)d_in[2];
  const float* w_q      = (const float*)d_in[3];
  const float* w_kv     = (const float*)d_in[4];
  const float* w_out    = (const float*)d_in[5];
  const float* inv_freq = (const float*)d_in[6];

  _Float16* ws = (_Float16*)d_ws;

  // 1) fused QKV projection + scale + RoPE (f16 outputs into workspace)
  qkv_rope_kernel<<<dim3(MTOT / 128, NKV / 128), 256, 0, stream>>>(
      x, w_q, w_kv, inv_freq, ws);

  // 2) per-head attention (8 heads per block, one wave each)
  attn_kernel<<<dim3(BB * S), 256, 0, stream>>>(
      pos_bias, focus, ws, ws + WS_O);

  // 3) output projection (f32 result)
  out_proj_kernel<<<dim3(MTOT / 128, HID / 128), 256, 0, stream>>>(
      ws + WS_O, w_out, (float*)d_out);
}
